// nestEGCN_85263690760752
// MI455X (gfx1250) — matrix operand order repaired and checked by reading, not compiled, not run on hardware
//
#include <hip/hip_runtime.h>
#include <hip/hip_bf16.h>

#define NN   25600
#define EE   409600
#define BBG  128
#define EFE  1024
#define DDIM 128
#define LLAY 8
#define KTOP 32
#define NPG  200

typedef __attribute__((ext_vector_type(16))) __bf16 v16bf;
typedef __attribute__((ext_vector_type(8)))  __bf16 v8bf;
typedef __attribute__((ext_vector_type(8)))  float  v8f;

__device__ __forceinline__ unsigned fkey(float f) {
  unsigned u = __float_as_uint(f);
  return (u & 0x80000000u) ? ~u : (u | 0x80000000u);
}
__device__ __forceinline__ float funkey(unsigned k) {
  unsigned u = (k & 0x80000000u) ? (k & 0x7fffffffu) : ~k;
  return __uint_as_float(u);
}

// ---------------------------------------------------------------------------
// Pack fp32 KxN (N=128) weight into WMMA B-operand layout, bf16.
// out[((nt*Ksteps+ks)*32+lane)*16+e] = W[(ks*32 + (e>>3)*16 + (lane>>4)*8 + (e&7))*128 + nt*16 + (lane&15)]
// ---------------------------------------------------------------------------
__global__ void pack_weight(const float* __restrict__ W, __bf16* __restrict__ out, int Ksteps) {
  int idx = blockIdx.x * blockDim.x + threadIdx.x;
  int total = Ksteps * 8 * 512;
  if (idx >= total) return;
  int e    = idx & 15;
  int lane = (idx >> 4) & 31;
  int grp  = idx >> 9;              // nt*Ksteps + ks
  int ks   = grp % Ksteps;
  int nt   = grp / Ksteps;
  int n = nt * 16 + (lane & 15);
  int k = ks * 32 + (e >> 3) * 16 + (lane >> 4) * 8 + (e & 7);
  out[idx] = (__bf16)W[(size_t)k * 128 + n];
}

// ---------------------------------------------------------------------------
// Generic WMMA GEMM: C_f32[M x 128] = A_bf16[M x (Ksteps*32)] @ Wpack (+bias, opt relu)
// one wave = 16-row tile, all 8 N-tiles (A reuse), f32 accumulate.
// ---------------------------------------------------------------------------
__global__ void gemm_wmma(const __bf16* __restrict__ A, const __bf16* __restrict__ Wp,
                          float* __restrict__ C, int M, int Ksteps,
                          const float* __restrict__ bias, int do_relu) {
  int wid = blockIdx.x * 8 + (threadIdx.x >> 5);
  if (wid >= (M >> 4)) return;
  int lane = threadIdx.x & 31;
  int col = lane & 15, half = lane >> 4;
  size_t lda = (size_t)Ksteps * 32;
  const __bf16* arow = A + (size_t)(wid * 16 + col) * lda;

  v8f acc[8];
  v8f zero = {0.f,0.f,0.f,0.f,0.f,0.f,0.f,0.f};
  for (int nt = 0; nt < 8; ++nt) acc[nt] = zero;

  for (int ks = 0; ks < Ksteps; ++ks) {
    union { v16bf v; v8bf h[2]; } a;
    a.h[0] = *(const v8bf*)(arow + ks * 32 + half * 8);
    a.h[1] = *(const v8bf*)(arow + ks * 32 + 16 + half * 8);
    for (int nt = 0; nt < 8; ++nt) {
      v16bf b = *(const v16bf*)(Wp + ((size_t)(nt * Ksteps + ks) * 32 + lane) * 16);
      acc[nt] = __builtin_amdgcn_wmma_f32_16x16x32_bf16(false, a.v, false, b,
                                                        (short)0, acc[nt], false, false);
    }
  }
  for (int nt = 0; nt < 8; ++nt) {
    int n = nt * 16 + col;
    float bv = bias ? bias[n] : 0.f;
    for (int r = 0; r < 8; ++r) {
      int m = r + half * 8;
      float v = acc[nt][r] + bv;
      if (do_relu) v = v > 0.f ? v : 0.f;
      C[(size_t)(wid * 16 + m) * 128 + n] = v;
    }
  }
}

// ---------------------------------------------------------------------------
// Fused edge GEMM: f_out = lrelu(e@Wfij + f_ni[src] + f_nj[dst] + bias, 0.01)
// + score = f_out . attn, + atomic segment-max(dst).
// ---------------------------------------------------------------------------
__global__ void edge_gemm_fused(const __bf16* __restrict__ Ebf, const __bf16* __restrict__ Wp,
                                const float* __restrict__ f_ni, const float* __restrict__ f_nj,
                                const int* __restrict__ src, const int* __restrict__ dst,
                                const float* __restrict__ bias, const float* __restrict__ attn,
                                __bf16* __restrict__ f_out, float* __restrict__ score,
                                unsigned* __restrict__ segmax) {
  int wid = blockIdx.x * 8 + (threadIdx.x >> 5);
  if (wid >= EE / 16) return;
  int lane = threadIdx.x & 31;
  int col = lane & 15, half = lane >> 4;
  int row0 = wid * 16;

  const __bf16* arow = Ebf + (size_t)(row0 + col) * 128;
  union { v16bf v; v8bf h[2]; } a[4];
  for (int ks = 0; ks < 4; ++ks) {
    a[ks].h[0] = *(const v8bf*)(arow + ks * 32 + half * 8);
    a[ks].h[1] = *(const v8bf*)(arow + ks * 32 + 16 + half * 8);
  }
  int se[8], de[8];
  for (int r = 0; r < 8; ++r) {
    int edge = row0 + r + half * 8;
    se[r] = src[edge]; de[r] = dst[edge];
  }
  float sp[8] = {0.f,0.f,0.f,0.f,0.f,0.f,0.f,0.f};
  v8f zero = {0.f,0.f,0.f,0.f,0.f,0.f,0.f,0.f};

  for (int nt = 0; nt < 8; ++nt) {
    v8f acc = zero;
    for (int ks = 0; ks < 4; ++ks) {
      v16bf b = *(const v16bf*)(Wp + ((size_t)(nt * 4 + ks) * 32 + lane) * 16);
      acc = __builtin_amdgcn_wmma_f32_16x16x32_bf16(false, a[ks].v, false, b,
                                                    (short)0, acc, false, false);
    }
    int n = nt * 16 + col;
    float bv = bias[n], at = attn[n];
    for (int r = 0; r < 8; ++r) {
      float v = acc[r] + f_ni[(size_t)se[r] * 128 + n] + f_nj[(size_t)de[r] * 128 + n] + bv;
      v = v > 0.f ? v : 0.01f * v;          // leaky_relu 0.01
      int edge = row0 + r + half * 8;
      f_out[(size_t)edge * 128 + n] = (__bf16)v;
      sp[r] += v * at;
    }
  }
  // reduce score partials across the 16 lanes of each half
  for (int m = 1; m < 16; m <<= 1)
    for (int r = 0; r < 8; ++r) sp[r] += __shfl_xor(sp[r], m, 32);
  if (col == 0) {
    for (int r = 0; r < 8; ++r) {
      int edge = row0 + r + half * 8;
      score[edge] = sp[r];
      atomicMax(&segmax[de[r]], fkey(sp[r]));
    }
  }
}

// ---------------------------------------------------------------------------
__global__ void softmax_norm(const float* __restrict__ score, const int* __restrict__ dst,
                             const unsigned* __restrict__ segmax, float* __restrict__ exv,
                             float* __restrict__ segsum, int nE) {
  int e = blockIdx.x * blockDim.x + threadIdx.x;
  if (e >= nE) return;
  float m = funkey(segmax[dst[e]]);
  float ex = __expf(score[e] - m);
  exv[e] = ex;
  atomicAdd(&segsum[dst[e]], ex);
}

__global__ void aggregate_edges(const float* __restrict__ hs, const int* __restrict__ src,
                                const int* __restrict__ dst, const float* __restrict__ exv,
                                const float* __restrict__ segsum, float* __restrict__ hnew) {
  size_t idx = (size_t)blockIdx.x * blockDim.x + threadIdx.x;
  if (idx >= (size_t)EE * 128) return;
  int e = (int)(idx >> 7), d = (int)(idx & 127);
  float coef = exv[e] / segsum[dst[e]];
  atomicAdd(&hnew[(size_t)dst[e] * 128 + d], hs[(size_t)src[e] * 128 + d] * coef);
}

__global__ void finish_layer(const float* __restrict__ hnew, __bf16* __restrict__ hbf) {
  size_t idx = (size_t)blockIdx.x * blockDim.x + threadIdx.x;
  if (idx >= (size_t)NN * 128) return;
  float v = hnew[idx];
  hbf[idx] = (__bf16)(v > 0.f ? v : 0.f);
}

// ---------------------------------------------------------------------------
__global__ void embed_h(const int* __restrict__ tok, const float* __restrict__ emb,
                        __bf16* __restrict__ hbf) {
  size_t idx = (size_t)blockIdx.x * blockDim.x + threadIdx.x;
  if (idx >= (size_t)NN * 128) return;
  int n = (int)(idx >> 7), d = (int)(idx & 127);
  float v = emb[(size_t)tok[n] * 128 + d];
  hbf[idx] = (__bf16)(v > 0.f ? v : 0.f);
}

__global__ void embed_e(const int* __restrict__ tok, const float* __restrict__ emb,
                        __bf16* __restrict__ ebf) {
  size_t idx = (size_t)blockIdx.x * blockDim.x + threadIdx.x;
  if (idx >= (size_t)EE * 128) return;
  int e = (int)(idx >> 7), d = (int)(idx & 127);
  ebf[idx] = (__bf16)emb[(size_t)tok[e] * 128 + d];
}

// ---------------------------------------------------------------------------
// Sort each node's 128 features ascending (one 128-thread block / node).
__global__ void sort_node(const float* __restrict__ h2, float* __restrict__ hsort) {
  __shared__ float s[128];
  int n = blockIdx.x, t = threadIdx.x;
  s[t] = h2[(size_t)n * 128 + t];
  __syncthreads();
  for (int k = 2; k <= 128; k <<= 1)
    for (int j = k >> 1; j > 0; j >>= 1) {
      int ixj = t ^ j;
      if (ixj > t) {
        float a = s[t], b = s[ixj];
        bool up = ((t & k) == 0);
        if ((a > b) == up) { s[t] = b; s[ixj] = a; }
      }
      __syncthreads();
    }
  hsort[(size_t)n * 128 + t] = s[t];
}

// Per graph: top-32 nodes by last sorted channel; emit pooled (B x 4096) bf16.
__global__ void sortpool_topk(const float* __restrict__ hsort, __bf16* __restrict__ pooled) {
  int b = blockIdx.x, t = threadIdx.x;           // 256 threads
  __shared__ float vals[256];
  __shared__ float rv[256];
  __shared__ int   ri[256];
  __shared__ int   sel[KTOP];
  vals[t] = (t < NPG) ? hsort[((size_t)b * NPG + t) * 128 + 127] : -__builtin_inff();
  __syncthreads();
  for (int k = 0; k < KTOP; ++k) {
    rv[t] = vals[t]; ri[t] = t;
    __syncthreads();
    for (int s = 128; s > 0; s >>= 1) {
      if (t < s) {
        if (rv[t + s] > rv[t] || (rv[t + s] == rv[t] && ri[t + s] < ri[t])) {
          rv[t] = rv[t + s]; ri[t] = ri[t + s];
        }
      }
      __syncthreads();
    }
    if (t == 0) { sel[k] = ri[0]; vals[ri[0]] = -__builtin_inff(); }
    __syncthreads();
  }
  for (int i = t; i < KTOP * 128; i += 256) {
    int k = i >> 7, d = i & 127;
    pooled[(size_t)b * 4096 + i] = (__bf16)hsort[((size_t)b * NPG + sel[k]) * 128 + d];
  }
}

// ---------------------------------------------------------------------------
__global__ void ft_scores(const float* __restrict__ ft, const float* __restrict__ al3,
                          const float* __restrict__ ar3, float* __restrict__ alv,
                          float* __restrict__ arv) {
  int b = threadIdx.x;                            // 128 threads
  float sa = 0.f, sr = 0.f;
  for (int d = 0; d < 128; ++d) { float v = ft[b * 128 + d]; sa += v * al3[d]; sr += v * ar3[d]; }
  alv[b] = sa; arv[b] = sr;
}

__global__ void fg_pass1(const int* __restrict__ fs, const int* __restrict__ fd,
                         const float* __restrict__ alv, const float* __restrict__ arv,
                         float* __restrict__ sc, unsigned* __restrict__ smax) {
  int e = blockIdx.x * blockDim.x + threadIdx.x;
  if (e >= EFE) return;
  float v = alv[fs[e]] + arv[fd[e]];
  v = v > 0.f ? v : 0.2f * v;                     // leaky_relu 0.2
  sc[e] = v;
  atomicMax(&smax[fd[e]], fkey(v));
}

__global__ void fg_aggregate(const float* __restrict__ ft, const int* __restrict__ fs,
                             const int* __restrict__ fd, const float* __restrict__ exv,
                             const float* __restrict__ ssum, float* __restrict__ g0) {
  int idx = blockIdx.x * blockDim.x + threadIdx.x;
  if (idx >= EFE * 128) return;
  int e = idx >> 7, d = idx & 127;
  float coef = exv[e] / ssum[fd[e]];
  atomicAdd(&g0[(size_t)fd[e] * 128 + d], ft[(size_t)fs[e] * 128 + d] * coef);
}

__global__ void g_bias_relu(const float* __restrict__ g0, const float* __restrict__ b3,
                            __bf16* __restrict__ gbf) {
  int idx = blockIdx.x * blockDim.x + threadIdx.x;
  if (idx >= BBG * 128) return;
  float v = g0[idx] + b3[idx & 127];
  gbf[idx] = (__bf16)(v > 0.f ? v : 0.f);
}

__global__ void final_out(const float* __restrict__ g2, const float* __restrict__ Wc,
                          const float* __restrict__ bc, float* __restrict__ out) {
  int idx = threadIdx.x;                          // 256 threads
  int b = idx >> 1, c = idx & 1;
  float s = bc[c];
  for (int k = 0; k < 128; ++k) s += g2[(size_t)b * 128 + k] * Wc[k * 2 + c];
  out[idx] = s;
}

// ---------------------------------------------------------------------------
extern "C" void kernel_launch(void* const* d_in, const int* in_sizes, int n_in,
                              void* d_out, int out_size, void* d_ws, size_t ws_size,
                              hipStream_t stream) {
  const int*   tokens_h   = (const int*)d_in[0];
  const int*   tokens_e   = (const int*)d_in[1];
  const int*   src        = (const int*)d_in[2];
  const int*   dst        = (const int*)d_in[3];
  const int*   fg_src     = (const int*)d_in[4];
  const int*   fg_dst     = (const int*)d_in[5];
  const float* token_emb  = (const float*)d_in[6];
  const float* e_tok_emb  = (const float*)d_in[7];
  const float* W_ni       = (const float*)d_in[8];
  const float* W_nj       = (const float*)d_in[9];
  const float* W_fij      = (const float*)d_in[10];
  const float* W_node     = (const float*)d_in[11];
  const float* attn_e     = (const float*)d_in[12];
  const float* bias_e     = (const float*)d_in[13];
  const float* Wf         = (const float*)d_in[14];
  const float* bf_        = (const float*)d_in[15];
  const float* W3         = (const float*)d_in[16];
  const float* al3        = (const float*)d_in[17];
  const float* ar3        = (const float*)d_in[18];
  const float* b3         = (const float*)d_in[19];
  const float* Wl         = (const float*)d_in[20];
  const float* bl         = (const float*)d_in[21];
  const float* Wc         = (const float*)d_in[22];
  const float* bc         = (const float*)d_in[23];

  size_t off = 0;
  char* base = (char*)d_ws;
  auto alloc = [&](size_t bytes) -> char* {
    char* p = base + off;
    off += (bytes + 255) & ~(size_t)255;
    return p;
  };

  __bf16*   h_bf   = (__bf16*)alloc((size_t)NN * 128 * 2);
  __bf16*   e_a    = (__bf16*)alloc((size_t)EE * 128 * 2);
  __bf16*   e_b    = (__bf16*)alloc((size_t)EE * 128 * 2);
  float*    f_ni   = (float*)alloc((size_t)NN * 128 * 4);
  float*    f_nj   = (float*)alloc((size_t)NN * 128 * 4);
  float*    hs     = (float*)alloc((size_t)NN * 128 * 4);
  float*    hnew   = (float*)alloc((size_t)NN * 128 * 4);
  float*    score  = (float*)alloc((size_t)EE * 4);
  float*    exv    = (float*)alloc((size_t)EE * 4);
  unsigned* segmax = (unsigned*)alloc((size_t)NN * 4);
  float*    segsum = (float*)alloc((size_t)NN * 4);
  __bf16*   pWni   = (__bf16*)alloc((size_t)LLAY * 16384 * 2);
  __bf16*   pWnj   = (__bf16*)alloc((size_t)LLAY * 16384 * 2);
  __bf16*   pWfij  = (__bf16*)alloc((size_t)LLAY * 16384 * 2);
  __bf16*   pWnode = (__bf16*)alloc((size_t)LLAY * 16384 * 2);
  __bf16*   pWf    = (__bf16*)alloc((size_t)16384 * 2);
  __bf16*   pW3    = (__bf16*)alloc((size_t)524288 * 2);
  __bf16*   pWl    = (__bf16*)alloc((size_t)16384 * 2);
  __bf16*   pooled = (__bf16*)alloc((size_t)BBG * 4096 * 2);
  float*    ft     = (float*)alloc((size_t)BBG * 128 * 4);
  float*    alv    = (float*)alloc((size_t)BBG * 4);
  float*    arv    = (float*)alloc((size_t)BBG * 4);
  float*    sc2    = (float*)alloc((size_t)EFE * 4);
  float*    ex2    = (float*)alloc((size_t)EFE * 4);
  unsigned* smax2  = (unsigned*)alloc((size_t)BBG * 4);
  float*    ssum2  = (float*)alloc((size_t)BBG * 4);
  float*    g0     = (float*)alloc((size_t)BBG * 128 * 4);
  __bf16*   g_bf   = (__bf16*)alloc((size_t)BBG * 128 * 2);
  float*    g2     = (float*)alloc((size_t)BBG * 128 * 4);
  // dead-after-layers buffers reused:
  float* h2    = f_ni;   // N x 128 post-Wf
  float* hsort = f_nj;   // N x 128 sorted

  // --- pack all weights to WMMA B-layout (bf16) ---
  for (int l = 0; l < LLAY; ++l) {
    pack_weight<<<64, 256, 0, stream>>>(W_ni   + (size_t)l * 16384, pWni   + (size_t)l * 16384, 4);
    pack_weight<<<64, 256, 0, stream>>>(W_nj   + (size_t)l * 16384, pWnj   + (size_t)l * 16384, 4);
    pack_weight<<<64, 256, 0, stream>>>(W_fij  + (size_t)l * 16384, pWfij  + (size_t)l * 16384, 4);
    pack_weight<<<64, 256, 0, stream>>>(W_node + (size_t)l * 16384, pWnode + (size_t)l * 16384, 4);
  }
  pack_weight<<<64, 256, 0, stream>>>(Wf, pWf, 4);
  pack_weight<<<2048, 256, 0, stream>>>(W3, pW3, 128);
  pack_weight<<<64, 256, 0, stream>>>(Wl, pWl, 4);

  // --- embeddings ---
  embed_h<<<(NN * 128) / 256, 256, 0, stream>>>(tokens_h, token_emb, h_bf);
  embed_e<<<((size_t)EE * 128) / 256, 256, 0, stream>>>(tokens_e, e_tok_emb, e_a);

  __bf16* e_cur = e_a;
  __bf16* e_nxt = e_b;

  // --- 8 EGATConv layers ---
  for (int l = 0; l < LLAY; ++l) {
    hipMemsetAsync(hnew, 0, (size_t)NN * 128 * 4, stream);
    hipMemsetAsync(segmax, 0, (size_t)NN * 4, stream);
    hipMemsetAsync(segsum, 0, (size_t)NN * 4, stream);

    gemm_wmma<<<NN / 128, 256, 0, stream>>>(h_bf, pWni   + (size_t)l * 16384, f_ni, NN, 4, nullptr, 0);
    gemm_wmma<<<NN / 128, 256, 0, stream>>>(h_bf, pWnj   + (size_t)l * 16384, f_nj, NN, 4, nullptr, 0);
    gemm_wmma<<<NN / 128, 256, 0, stream>>>(h_bf, pWnode + (size_t)l * 16384, hs,   NN, 4, nullptr, 0);

    edge_gemm_fused<<<EE / 128, 256, 0, stream>>>(e_cur, pWfij + (size_t)l * 16384,
                                                  f_ni, f_nj, src, dst,
                                                  bias_e + l * 128, attn_e + l * 128,
                                                  e_nxt, score, segmax);
    softmax_norm<<<EE / 256, 256, 0, stream>>>(score, dst, segmax, exv, segsum, EE);
    aggregate_edges<<<((size_t)EE * 128) / 256, 256, 0, stream>>>(hs, src, dst, exv, segsum, hnew);
    finish_layer<<<(NN * 128) / 256, 256, 0, stream>>>(hnew, h_bf);

    __bf16* t = e_cur; e_cur = e_nxt; e_nxt = t;
  }

  // --- linear_forward + SortPooling ---
  gemm_wmma<<<NN / 128, 256, 0, stream>>>(h_bf, pWf, h2, NN, 4, bf_, 1);
  sort_node<<<NN, 128, 0, stream>>>(h2, hsort);
  sortpool_topk<<<BBG, 256, 0, stream>>>(hsort, pooled);

  // --- GATConv on final graph ---
  gemm_wmma<<<1, 256, 0, stream>>>(pooled, pW3, ft, BBG, 128, nullptr, 0);
  ft_scores<<<1, 128, 0, stream>>>(ft, al3, ar3, alv, arv);

  hipMemsetAsync(g0, 0, (size_t)BBG * 128 * 4, stream);
  hipMemsetAsync(smax2, 0, (size_t)BBG * 4, stream);
  hipMemsetAsync(ssum2, 0, (size_t)BBG * 4, stream);

  fg_pass1<<<EFE / 256, 256, 0, stream>>>(fg_src, fg_dst, alv, arv, sc2, smax2);
  softmax_norm<<<EFE / 256, 256, 0, stream>>>(sc2, fg_dst, smax2, ex2, ssum2, EFE);
  fg_aggregate<<<(EFE * 128) / 256, 256, 0, stream>>>(ft, fg_src, fg_dst, ex2, ssum2, g0);
  g_bias_relu<<<(BBG * 128) / 256, 256, 0, stream>>>(g0, b3, g_bf);

  // --- final dense layers ---
  gemm_wmma<<<1, 256, 0, stream>>>(g_bf, pWl, g2, BBG, 4, bl, 1);
  final_out<<<1, 256, 0, stream>>>(g2, Wc, bc, (float*)d_out);
}
